// StandardAttention_36447092474048
// MI455X (gfx1250) — matrix operand order repaired and checked by reading, not compiled, not run on hardware
//
#include <hip/hip_runtime.h>
#include <hip/hip_bf16.h>
#include <math.h>

typedef __bf16 bf16_t;
typedef __attribute__((ext_vector_type(16))) __bf16 v16bf;
typedef __attribute__((ext_vector_type(8)))  __bf16 v8bf;
typedef __attribute__((ext_vector_type(4)))  __bf16 v4bf;
typedef __attribute__((ext_vector_type(8)))  float  v8f;
typedef __attribute__((ext_vector_type(4)))  float  v4f;

#define DIM       1024
#define HEADS     16
#define HEAD_DIM  64
#define SEQ       2048
#define BATCH     4
#define TOKENS    (BATCH * SEQ)   /* 8192 */
#define BH        (BATCH * HEADS) /* 64   */
#define QKV_N     (3 * DIM)       /* 3072 */
#define ATTN_SCALE 0.125f         /* 1/sqrt(64) */

static __device__ __forceinline__ v8bf cvt8(v8f v) {
  return __builtin_convertvector(v, v8bf);
}
static __device__ __forceinline__ bf16_t cvt1(float f) { return (bf16_t)f; }

// A-matrix fragment (16x32, MxK) from row-major [row][k], ld in elements.
// Lane L: row L%16; elems 0..7 -> k = 8*(L/16)+e, elems 8..15 -> k = 16+8*(L/16)+(e-8).
static __device__ __forceinline__ v16bf load_frag_a(const bf16_t* __restrict__ src,
                                                    int ld, int lane) {
  const int row  = lane & 15;
  const int koff = (lane >> 4) << 3; // 0 or 8
  const bf16_t* p = src + row * ld + koff;
  v8bf lo = *(const v8bf*)(p);
  v8bf hi = *(const v8bf*)(p + 16);
  v16bf r;
#pragma unroll
  for (int i = 0; i < 8; ++i) { r[i] = lo[i]; r[i + 8] = hi[i]; }
  return r;
}

// B-matrix fragment (32x16, KxN) with B[k][n] = src[n][k] (src row-major, ld elems).
// Lane L: column n = L%16; elems 0..7 -> k = 8*(L/16)+e, elems 8..15 -> k = 16+8*(L/16)+(e-8).
static __device__ __forceinline__ v16bf load_frag_b(const bf16_t* __restrict__ src,
                                                    int ld, int lane) {
  const int row  = lane & 15;
  const int koff = (lane >> 4) << 3; // 0 or 8
  const bf16_t* p = src + row * ld + koff;
  v8bf lo = *(const v8bf*)(p);
  v8bf hi = *(const v8bf*)(p + 16);
  v16bf r;
#pragma unroll
  for (int i = 0; i < 8; ++i) { r[i] = lo[i]; r[i + 8] = hi[i]; }
  return r;
}

static __device__ __forceinline__ v8f wmma_bf16(v16bf a, v16bf b, v8f c) {
  return __builtin_amdgcn_wmma_f32_16x16x32_bf16(false, a, false, b, (short)0, c,
                                                 false, false);
}

// Async global->LDS copy, 16B per lane (512B per wave per instruction). ASYNCcnt.
static __device__ __forceinline__ void async_ld16(unsigned lds_off, const void* gptr) {
  asm volatile("global_load_async_to_lds_b128 %0, %1, off"
               :: "v"(lds_off), "v"(gptr)
               : "memory");
}

// ---------------------------------------------------------------- conversions
__global__ void cvt_f32_bf16_v4(const float* __restrict__ src,
                                bf16_t* __restrict__ dst, int n4) {
  int i = blockIdx.x * blockDim.x + threadIdx.x;
  if (i < n4) {
    v4f f = ((const v4f*)src)[i];
    ((v4bf*)dst)[i] = __builtin_convertvector(f, v4bf);
  }
}

// ------------------------------------------------------------------ QKV GEMM
// A = w_qkv (M = 3072 features), B = x (N = 8192 tokens).
// C fragment: lane's 8 accumulators = 8 consecutive features -> packed stores
// for q/k ([bh][tok][d], contiguous in d); v transposed ([bh][d][tok]) scatters.
__global__ void __launch_bounds__(128)
qkv_gemm_kernel(const bf16_t* __restrict__ xb, const bf16_t* __restrict__ wqb,
                bf16_t* __restrict__ qb, bf16_t* __restrict__ kb,
                bf16_t* __restrict__ vtb) {
  const int lane = threadIdx.x & 31;
  const int wid  = blockIdx.x * (blockDim.x >> 5) + (threadIdx.x >> 5);
  const int TT   = TOKENS / 64; // 128 token tiles
  const int ft   = wid / TT;    // feature tile (32 features)
  const int tt   = wid % TT;    // token tile (64 tokens)

  const bf16_t* A = wqb + (size_t)(ft * 32) * DIM; // feature rows
  const bf16_t* B = xb  + (size_t)(tt * 64) * DIM; // token rows

  v8f acc[2][4] = {};

  for (int kk = 0; kk < DIM; kk += 32) {
    __builtin_prefetch(A + kk + 32, 0, 3);
    __builtin_prefetch(B + kk + 32, 0, 3);
    v16bf a0 = load_frag_a(A + kk, DIM, lane);
    v16bf a1 = load_frag_a(A + 16 * DIM + kk, DIM, lane);
    v16bf bfr[4];
#pragma unroll
    for (int j = 0; j < 4; ++j)
      bfr[j] = load_frag_b(B + (size_t)(16 * j) * DIM + kk, DIM, lane);
#pragma unroll
    for (int j = 0; j < 4; ++j) {
      acc[0][j] = wmma_bf16(a0, bfr[j], acc[0][j]);
      acc[1][j] = wmma_bf16(a1, bfr[j], acc[1][j]);
    }
  }

  const int col0 = lane & 15;
  const int half = lane >> 4;
  // Tile-uniform: 32-wide aligned feature tile -> one head, one of q/k/v;
  // 64-wide aligned token tile -> one batch.
  const int nbase = ft * 32;
  const int sel   = nbase >> 10;                 // 0=q 1=k 2=v
  const int head  = (nbase & (DIM - 1)) >> 6;
  const int bI    = (tt * 64) >> 11;
  const int bh    = bI * HEADS + head;

#pragma unroll
  for (int i = 0; i < 2; ++i) {
    const int d0 = (nbase & 63) + i * 16 + 8 * half; // 8 consecutive d from here
#pragma unroll
    for (int j = 0; j < 4; ++j) {
      const int tok = (tt * 64 + j * 16 + col0) & (SEQ - 1);
      v8bf pk = cvt8(acc[i][j]);
      if (sel == 0) {
        *(v8bf*)&qb[((size_t)bh * SEQ + tok) * HEAD_DIM + d0] = pk;
      } else if (sel == 1) {
        *(v8bf*)&kb[((size_t)bh * SEQ + tok) * HEAD_DIM + d0] = pk;
      } else {
#pragma unroll
        for (int r = 0; r < 8; ++r)
          vtb[((size_t)bh * HEAD_DIM + d0 + r) * SEQ + tok] = pk[r];
      }
    }
  }
}

// ------------------------------------------------------------- attention
// 4 waves x 16 query rows per workgroup. K/V blocks (32 keys) staged in LDS
// via double-buffered GLOBAL_LOAD_ASYNC_TO_LDS_B128; flash-attention softmax.
__global__ void __launch_bounds__(128)
attn_kernel(const bf16_t* __restrict__ qb, const bf16_t* __restrict__ kbuf,
            const bf16_t* __restrict__ vtb, bf16_t* __restrict__ ao) {
  __shared__ bf16_t ktile[2][32 * 64] __attribute__((aligned(16))); // [buf][n'][d]
  __shared__ bf16_t vtile[2][64 * 32] __attribute__((aligned(16))); // [buf][d][n']
  __shared__ bf16_t psh[4][16][32]    __attribute__((aligned(16)));

  const int lane = threadIdx.x & 31;
  const int w    = threadIdx.x >> 5;
  const int half = lane >> 4;
  const int col0 = lane & 15;
  const int bh   = blockIdx.y;
  const int m0   = blockIdx.x * 64 + w * 16;

  const char* Kg = (const char*)(kbuf + (size_t)bh * SEQ * HEAD_DIM);
  const char* Vg = (const char*)(vtb  + (size_t)bh * HEAD_DIM * SEQ);

  // Issue this wave's share (4 async b128 ops) of the K+V block at `nb`.
  auto issue = [&](int c, int nb) {
    const unsigned kbase = (unsigned)(size_t)&ktile[c][0];
    const unsigned vbase = (unsigned)(size_t)&vtile[c][0];
#pragma unroll
    for (int s = 0; s < 2; ++s) {
      unsigned o = (unsigned)(w * 1024 + s * 512 + lane * 16); // 0..4080
      // K block: 32 rows x 128B, contiguous at Kg + nb*128.
      async_ld16(kbase + o, Kg + (size_t)nb * 128 + o);
      // V block: 64 rows (d) x 64B slices of rows with stride SEQ*2.
      async_ld16(vbase + o,
                 Vg + (size_t)(o >> 6) * (SEQ * 2) + (o & 63) + (size_t)nb * 2);
    }
  };

  const bf16_t* Q = qb + ((size_t)bh * SEQ + m0) * HEAD_DIM;
  v16bf qa0 = load_frag_a(Q, HEAD_DIM, lane);
  v16bf qa1 = load_frag_a(Q + 32, HEAD_DIM, lane);

  float rm[8], rl[8];
#pragma unroll
  for (int r = 0; r < 8; ++r) { rm[r] = -1e30f; rl[r] = 0.0f; }
  v8f oacc[4] = {};

  issue(0, 0);

  for (int nb = 0; nb < SEQ; nb += 32) {
    const int c = (nb >> 5) & 1;
    if (nb + 32 < SEQ) {
      issue(c ^ 1, nb + 32);
      asm volatile("s_wait_asynccnt 0x4" ::: "memory"); // current block landed
    } else {
      asm volatile("s_wait_asynccnt 0x0" ::: "memory");
    }
    __syncthreads(); // publish current K/V block to all waves

    // S = Q K^T over two 16-column sub-tiles (reads LDS via ds_load_b128)
    v8f s[2];
#pragma unroll
    for (int t = 0; t < 2; ++t) {
      const bf16_t* Kt = &ktile[c][(size_t)(16 * t) * HEAD_DIM];
      v8f cacc = {};
      cacc = wmma_bf16(qa0, load_frag_b(Kt, HEAD_DIM, lane), cacc);
      cacc = wmma_bf16(qa1, load_frag_b(Kt + 32, HEAD_DIM, lane), cacc);
      s[t] = cacc * ATTN_SCALE;
    }

    // online softmax; row reductions stay within each 16-lane half (wave32)
    float esc[8];
#pragma unroll
    for (int r = 0; r < 8; ++r) {
      float cmax = fmaxf(s[0][r], s[1][r]);
#pragma unroll
      for (int mask = 1; mask <= 8; mask <<= 1)
        cmax = fmaxf(cmax, __shfl_xor(cmax, mask, 32));
      float mnew = fmaxf(rm[r], cmax);
      float p0 = __expf(s[0][r] - mnew);
      float p1 = __expf(s[1][r] - mnew);
      float rs = p0 + p1;
#pragma unroll
      for (int mask = 1; mask <= 8; mask <<= 1)
        rs += __shfl_xor(rs, mask, 32);
      esc[r] = __expf(rm[r] - mnew);
      rl[r] = rl[r] * esc[r] + rs;
      rm[r] = mnew;
      psh[w][r + 8 * half][col0]      = cvt1(p0);
      psh[w][r + 8 * half][16 + col0] = cvt1(p1);
    }
#pragma unroll
    for (int j = 0; j < 4; ++j)
#pragma unroll
      for (int r = 0; r < 8; ++r) oacc[j][r] *= esc[r];

    // P as A operand (wave-private LDS; same-wave DS ops are in-order)
    v16bf pa = load_frag_a(&psh[w][0][0], 32, lane);
#pragma unroll
    for (int j = 0; j < 4; ++j) {
      v16bf vb = load_frag_b(&vtile[c][(size_t)(16 * j) * 32], 32, lane);
      oacc[j] = wmma_bf16(pa, vb, oacc[j]);
    }
    __syncthreads(); // all waves done reading buffers before next overwrite
  }

  float rinv[8];
#pragma unroll
  for (int r = 0; r < 8; ++r) rinv[r] = 1.0f / rl[r];

  const int bI = bh / HEADS, head = bh % HEADS;
#pragma unroll
  for (int j = 0; j < 4; ++j) {
#pragma unroll
    for (int r = 0; r < 8; ++r) {
      int tok = m0 + r + 8 * half;
      int col = head * HEAD_DIM + 16 * j + col0;
      ao[((size_t)bI * SEQ + tok) * DIM + col] = cvt1(oacc[j][r] * rinv[r]);
    }
  }
}

// ------------------------------------------------------------- output GEMM
// A = w_out (M = 1024 features), B = ao (N = 8192 tokens).
// Lane's 8 accumulators = 8 consecutive output features of one token ->
// one 32B f32 store per fragment; bias added as v8f.
__global__ void __launch_bounds__(128)
out_gemm_kernel(const bf16_t* __restrict__ ao, const bf16_t* __restrict__ wob,
                const float* __restrict__ bias, float* __restrict__ out) {
  const int lane = threadIdx.x & 31;
  const int wid  = blockIdx.x * (blockDim.x >> 5) + (threadIdx.x >> 5);
  const int TT   = TOKENS / 64; // 128 token tiles
  const int ft   = wid / TT;    // feature tile (32 features)
  const int tt   = wid % TT;    // token tile (64 tokens)

  const bf16_t* A = wob + (size_t)(ft * 32) * DIM;
  const bf16_t* B = ao  + (size_t)(tt * 64) * DIM;

  v8f acc[2][4] = {};

  for (int kk = 0; kk < DIM; kk += 32) {
    __builtin_prefetch(A + kk + 32, 0, 3);
    __builtin_prefetch(B + kk + 32, 0, 3);
    v16bf a0 = load_frag_a(A + kk, DIM, lane);
    v16bf a1 = load_frag_a(A + 16 * DIM + kk, DIM, lane);
    v16bf bfr[4];
#pragma unroll
    for (int j = 0; j < 4; ++j)
      bfr[j] = load_frag_b(B + (size_t)(16 * j) * DIM + kk, DIM, lane);
#pragma unroll
    for (int j = 0; j < 4; ++j) {
      acc[0][j] = wmma_bf16(a0, bfr[j], acc[0][j]);
      acc[1][j] = wmma_bf16(a1, bfr[j], acc[1][j]);
    }
  }

  const int col0 = lane & 15;
  const int half = lane >> 4;
#pragma unroll
  for (int i = 0; i < 2; ++i) {
    const int n0 = ft * 32 + i * 16 + 8 * half; // 8 consecutive output features
    const v8f bv = *(const v8f*)&bias[n0];
#pragma unroll
    for (int j = 0; j < 4; ++j) {
      const int tok = tt * 64 + j * 16 + col0;
      *(v8f*)&out[(size_t)tok * DIM + n0] = acc[i][j] + bv;
    }
  }
}

// ---------------------------------------------------------------- launcher
extern "C" void kernel_launch(void* const* d_in, const int* in_sizes, int n_in,
                              void* d_out, int out_size, void* d_ws, size_t ws_size,
                              hipStream_t stream) {
  (void)in_sizes; (void)n_in; (void)out_size; (void)ws_size;
  const float* x     = (const float*)d_in[0];
  const float* w_qkv = (const float*)d_in[1];
  const float* w_out = (const float*)d_in[2];
  const float* b_out = (const float*)d_in[3];
  float* out = (float*)d_out;

  char* ws = (char*)d_ws;
  bf16_t* xb  = (bf16_t*)(ws + 0);         // 8192*1024*2  = 16 MB
  bf16_t* wqb = (bf16_t*)(ws + 16777216);  // 3072*1024*2  =  6 MB
  bf16_t* wob = (bf16_t*)(ws + 23068672);  // 1024*1024*2  =  2 MB
  bf16_t* qb  = (bf16_t*)(ws + 25165824);  // [bh][tok][d] = 16 MB
  bf16_t* kb  = (bf16_t*)(ws + 41943040);  // [bh][tok][d] = 16 MB
  bf16_t* vtb = (bf16_t*)(ws + 58720256);  // [bh][d][tok] = 16 MB
  bf16_t* ao  = (bf16_t*)(ws + 75497472);  // [tok][dim]   = 16 MB

  const int nx4 = TOKENS * DIM / 4;
  const int nq4 = QKV_N * DIM / 4;
  const int nw4 = DIM * DIM / 4;
  cvt_f32_bf16_v4<<<(nx4 + 255) / 256, 256, 0, stream>>>(x, xb, nx4);
  cvt_f32_bf16_v4<<<(nq4 + 255) / 256, 256, 0, stream>>>(w_qkv, wqb, nq4);
  cvt_f32_bf16_v4<<<(nw4 + 255) / 256, 256, 0, stream>>>(w_out, wob, nw4);

  // (3072/32)*(8192/64) = 12288 wave tiles, 4 waves per 128-thread block
  qkv_gemm_kernel<<<12288 / 4, 128, 0, stream>>>(xb, wqb, qb, kb, vtb);

  attn_kernel<<<dim3(SEQ / 64, BH), 128, 0, stream>>>(qb, kb, vtb, ao);

  // (1024/32)*(8192/64) = 4096 wave tiles
  out_gemm_kernel<<<4096 / 4, 128, 0, stream>>>(ao, wob, b_out, out);
}
